// ReservoirBlock_71047349010870
// MI455X (gfx1250) — matrix operand order repaired and checked by reading, not compiled, not run on hardware
//
#include <hip/hip_runtime.h>
#include <math.h>

// ---------------------------------------------------------------------------
// ReservoirBlock for MI455X (gfx1250, wave32, WMMA, TDM).
// Pipeline: LN -> xU GEMM -> sequential reservoir scan (persistent kernel,
// TDM-staged state in LDS, software grid barrier) -> LN -> MLP -> gated mix
// -> FFN.  All GEMMs use v_wmma_f32_16x16x32_bf16 (bf16 in, f32 accumulate).
// ---------------------------------------------------------------------------

#define BB 8
#define TT 2048
#define DD 768
#define RR 3072
#define MM (BB * TT)          // 16384 rows
#define SCAN_BLOCKS 24        // 24 blocks * 8 waves = 192 waves = R/16 n-tiles
#define LDS_STRIDE 3080       // 3072 + 8 elem pad (16B) -> stride 6160B

#if defined(__has_builtin)
#if __has_builtin(__builtin_amdgcn_tensor_load_to_lds)
#define HAVE_TDM 1
#endif
#endif
#ifndef HAVE_TDM
#define HAVE_TDM 0
#endif

typedef __attribute__((ext_vector_type(16))) __bf16 bf16x16;
typedef __attribute__((ext_vector_type(8)))  float  f32x8;
typedef unsigned int u32x4 __attribute__((ext_vector_type(4)));
typedef int          i32x8 __attribute__((ext_vector_type(8)));
typedef int          i32x4 __attribute__((ext_vector_type(4)));

union FragU { uint4 q[2]; bf16x16 v; };

__device__ __forceinline__ __bf16 f2bf(float f) {
  union { float f; unsigned u; } a; a.f = f;
  unsigned r = a.u + 0x7FFFu + ((a.u >> 16) & 1u);   // round-to-nearest-even
  unsigned short h = (unsigned short)(r >> 16);
  return __builtin_bit_cast(__bf16, h);
}
__device__ __forceinline__ float bf2f(__bf16 b) {
  unsigned short h = __builtin_bit_cast(unsigned short, b);
  union { unsigned u; float f; } a; a.u = ((unsigned)h) << 16;
  return a.f;
}
__device__ __forceinline__ float gelu_exact(float z) {
  return 0.5f * z * (1.0f + erff(z * 0.70710678118654752f));
}

// -------------------- weight convert + transpose (K x N f32 -> N x K bf16) --
__global__ void k_convert_T(const float* __restrict__ src, __bf16* __restrict__ dst,
                            int K, int N) {
  long total = (long)K * N;
  for (long i = (long)blockIdx.x * blockDim.x + threadIdx.x; i < total;
       i += (long)gridDim.x * blockDim.x) {
    long n = i / K;
    long k = i - n * K;
    dst[i] = f2bf(src[k * (long)N + n]);
  }
}

// -------------------- init: zero h state double-buffer + barrier ------------
__global__ void k_init(__bf16* h0, __bf16* h1, unsigned* bar) {
  int i = blockIdx.x * blockDim.x + threadIdx.x;
  if (i < 16 * RR) {
    unsigned short z = 0;
    h0[i] = __builtin_bit_cast(__bf16, z);
    h1[i] = __builtin_bit_cast(__bf16, z);
  }
  if (i < 4) bar[i] = 0u;
}

// -------------------- LayerNorm over (optionally concatenated) f32 row ------
__global__ void __launch_bounds__(256)
k_ln_f32(const float* __restrict__ p0, int d0,
         const float* __restrict__ p1, int d1,
         const float* __restrict__ w, const float* __restrict__ b,
         __bf16* __restrict__ out) {
  const int row = blockIdx.x;
  const int dim = d0 + d1;
  const float* r0 = p0 + (size_t)row * d0;
  const float* r1 = p1 ? (p1 + (size_t)row * d1) : nullptr;
  float s = 0.f, s2 = 0.f;
  for (int i = threadIdx.x; i < dim; i += 256) {
    float x = (i < d0) ? r0[i] : r1[i - d0];
    s += x; s2 += x * x;
  }
  __shared__ float red0[256], red1[256];
  red0[threadIdx.x] = s; red1[threadIdx.x] = s2;
  __syncthreads();
  for (int off = 128; off > 0; off >>= 1) {
    if (threadIdx.x < off) {
      red0[threadIdx.x] += red0[threadIdx.x + off];
      red1[threadIdx.x] += red1[threadIdx.x + off];
    }
    __syncthreads();
  }
  float mu = red0[0] / dim;
  float var = red1[0] / dim - mu * mu;
  float rs = rsqrtf(var + 1e-5f);
  __bf16* o = out + (size_t)row * dim;
  for (int i = threadIdx.x; i < dim; i += 256) {
    float x = (i < d0) ? r0[i] : r1[i - d0];
    o[i] = f2bf((x - mu) * rs * w[i] + b[i]);
  }
}

// -------------------- LayerNorm over bf16 row, in place ---------------------
__global__ void __launch_bounds__(256)
k_ln_bf16(__bf16* __restrict__ inout, const float* __restrict__ w,
          const float* __restrict__ b, int dim) {
  const int row = blockIdx.x;
  __bf16* p = inout + (size_t)row * dim;
  float s = 0.f, s2 = 0.f;
  for (int i = threadIdx.x; i < dim; i += 256) {
    float x = bf2f(p[i]);
    s += x; s2 += x * x;
  }
  __shared__ float red0[256], red1[256];
  red0[threadIdx.x] = s; red1[threadIdx.x] = s2;
  __syncthreads();
  for (int off = 128; off > 0; off >>= 1) {
    if (threadIdx.x < off) {
      red0[threadIdx.x] += red0[threadIdx.x + off];
      red1[threadIdx.x] += red1[threadIdx.x + off];
    }
    __syncthreads();
  }
  float mu = red0[0] / dim;
  float var = red1[0] / dim - mu * mu;
  float rs = rsqrtf(var + 1e-5f);
  for (int i = threadIdx.x; i < dim; i += 256) {
    float x = bf2f(p[i]);
    p[i] = f2bf((x - mu) * rs * w[i] + b[i]);
  }
}

// -------------------- generic WMMA GEMM with fused epilogues ----------------
// C(M x N) = epilogue( A(M x K, bf16 row-major) @ WT(N x K, bf16 row-major)^T )
// Block: 256 thr = 8 waves arranged 4(M) x 2(N) -> 64 x 32 output per block.
// A frag (ISA 16-bit A 16x32): lane lh=0 -> K {0..7, 16..23}; lh=1 -> {8..15, 24..31}
// B frag (ISA 16-bit B 32x16): lane lh=0 -> K {0..7, 16..23}; lh=1 -> {8..15, 24..31}
// MODE 0: outf = z                     (xU, no bias)
// MODE 1: outb = bf16(gelu(z + bias))
// MODE 2: outf = exp(*sp) * (z + bias)
// MODE 3: g = sigmoid(z+bias); outf = g*aux0 + (1-g)*aux1   (gated mix)
// MODE 4: outf = aux0 + (*sp) * (z + bias)                  (final residual)
template <int MODE>
__global__ void __launch_bounds__(256)
k_gemm(const __bf16* __restrict__ A, const __bf16* __restrict__ WT,
       const float* __restrict__ bias, int K, int N,
       __bf16* __restrict__ outb, float* __restrict__ outf,
       const float* __restrict__ aux0, const float* __restrict__ aux1,
       const float* __restrict__ sp) {
  const int lane = threadIdx.x & 31;
  const int wave = threadIdx.x >> 5;
  const int mt = wave & 3, nt = wave >> 2;
  const int row0 = blockIdx.x * 64 + mt * 16;
  const int col0 = blockIdx.y * 32 + nt * 16;
  const int lm = lane & 15;
  const int lh = (lane >> 4) & 1;

  const __bf16* ap = A + (size_t)(row0 + lm) * K + lh * 8;    // A layout
  const __bf16* bp = WT + (size_t)(col0 + lm) * K + lh * 8;   // B layout

  f32x8 acc = {};
  for (int kb = 0; kb < K; kb += 32) {
    FragU a, bb;
    a.q[0]  = *(const uint4*)(ap + kb);        // K: lh*8 .. +7
    a.q[1]  = *(const uint4*)(ap + kb + 16);   // K: 16+lh*8 .. +7
    bb.q[0] = *(const uint4*)(bp + kb);        // K: lh*8 .. +7
    bb.q[1] = *(const uint4*)(bp + kb + 16);   // K: 16+lh*8 .. +7
    if (kb + 256 < K) __builtin_prefetch(bp + kb + 256, 0, 0);
    acc = __builtin_amdgcn_wmma_f32_16x16x32_bf16(
        false, a.v, false, bb.v, (short)0, acc, false, false);
  }

  float scale = 1.f;
  if (MODE == 2) scale = expf(sp[0]);
  float alpha = 0.f;
  if (MODE == 4) alpha = sp[0];

#pragma unroll
  for (int v = 0; v < 8; ++v) {
    const int grow = row0 + v + 8 * lh;       // C layout: lanes 16-31 hold M+8
    const int gcol = col0 + lm;
    const size_t idx = (size_t)grow * N + gcol;
    float z = acc[v];
    if (MODE != 0) z += bias[gcol];
    if (MODE == 0) {
      outf[idx] = z;
    } else if (MODE == 1) {
      outb[idx] = f2bf(gelu_exact(z));
    } else if (MODE == 2) {
      outf[idx] = scale * z;
    } else if (MODE == 3) {
      float g = 1.f / (1.f + expf(-z));
      outf[idx] = g * aux0[idx] + (1.f - g) * aux1[idx];
    } else if (MODE == 4) {
      outf[idx] = aux0[idx] + alpha * z;
    }
  }
}

// -------------------- software grid barrier (release/acquire) ---------------
__device__ __forceinline__ void grid_barrier(unsigned* bar, unsigned nblk) {
  __syncthreads();
  if (threadIdx.x == 0) {
    __threadfence();
    volatile unsigned* vgen = bar + 1;
    unsigned g = *vgen;
    if (atomicAdd(bar, 1u) == nblk - 1u) {
      bar[0] = 0u;
      __threadfence();
      *vgen = g + 1u;
    } else {
      while (*vgen == g) { __builtin_amdgcn_s_sleep(1); }
      __threadfence();
    }
  }
  __syncthreads();
}

// -------------------- persistent reservoir scan -----------------------------
// h_t = tanh(xU_t + h_{t-1} @ W).  State is 16 x R bf16 (rows 8..15 zero pad)
// double-buffered in global memory; each step it is staged into LDS (96 KB)
// by ONE Tensor-Data-Mover DMA per block (iterated 1-row tiles with an LDS
// stride of 1540 dwords to reproduce the bank-conflict-free padded layout),
// and each of the 192 waves owns one 16-column slab of W^T (96 KB, L2/L0
// resident across all 2048 steps).
__global__ void __launch_bounds__(256)
k_scan(const float* __restrict__ xU, const __bf16* __restrict__ WTW,
       __bf16* __restrict__ rstates, __bf16* __restrict__ h0,
       __bf16* __restrict__ h1, unsigned* __restrict__ bar) {
  __shared__ alignas(16) unsigned short sh[16 * LDS_STRIDE];
  const int tid = threadIdx.x;
  const int lane = tid & 31;
  const int wave = tid >> 5;
  const int lm = lane & 15;
  const int lh = (lane >> 4) & 1;
  const int col = (blockIdx.x * 8 + wave) * 16 + lm;    // output column
  const __bf16* wp = WTW + (size_t)col * RR + lh * 8;   // B-layout K slice

  for (int t = 0; t < TT; ++t) {
    const __bf16* cur = (t & 1) ? h1 : h0;
    __bf16*       nxt = (t & 1) ? h0 : h1;

    // ---- stage current 16 x 3072 bf16 state into padded LDS ----
#if HAVE_TDM
    if (wave == 0) {
      unsigned lds_base = (unsigned)(size_t)(&sh[0]);
      unsigned long long ga = (unsigned long long)(size_t)cur;
      u32x4 g0;
      g0[0] = 1u;                                       // count=1 user desc
      g0[1] = lds_base;                                 // lds_addr (bytes)
      g0[2] = (unsigned)(ga & 0xFFFFFFFFu);             // global_addr[31:0]
      g0[3] = (unsigned)((ga >> 32) & 0x01FFFFFFu)      // global_addr[56:32]
              | (2u << 30);                             // type = 2 (image)
      i32x8 g1;
      g1[0] = (2 << 16) | (1 << 19);                    // data_size=4B, iterate
      g1[1] = (int)(1536u << 16);                       // tensor_dim0 = 1536 dw
      g1[2] = (int)(16u << 16);                         // tensor_dim1 = 16 rows
      g1[3] = (int)(1536u << 16);                       // tile_dim0 = 1536 dw
      g1[4] = 1;                                        // tile_dim1 = 1 row
      g1[5] = 1536;                                     // dim0_stride = 1536 dw
      g1[6] = 0;
      g1[7] = 0;
      i32x4 g2;
      g2[0] = 1;                                        // tensor_dim2 (unused)
      g2[1] = 1540;                                     // lds_addr_increment dw
      g2[2] = 1536;                                     // global_addr_incr dw
      g2[3] = (int)(15u << 16);                         // iterate_count-1 = 15
      i32x4 g3 = {0, 0, 0, 0};
      i32x8 g4 = {0, 0, 0, 0, 0, 0, 0, 0};              // 6-arg form: extra grp
      __builtin_amdgcn_tensor_load_to_lds(g0, g1, g2, g3, g4, 0);
      __builtin_amdgcn_s_wait_tensorcnt(0);
    }
    __syncthreads();
#else
    for (int i = tid; i < 16 * 384; i += 256) {
      int r = i / 384;
      int c = i - r * 384;
      unsigned ldsa = (unsigned)(size_t)(&sh[r * LDS_STRIDE + c * 8]);
      const __bf16* g = cur + (size_t)r * RR + c * 8;
      asm volatile("global_load_async_to_lds_b128 %0, %1, off"
                   :: "v"(ldsa), "v"(g) : "memory");
    }
    asm volatile("s_wait_asynccnt 0x0" ::: "memory");
    __syncthreads();
#endif

    f32x8 acc = {};
    const unsigned short* arow = &sh[lm * LDS_STRIDE + lh * 8];
    for (int kb = 0; kb < RR; kb += 32) {
      FragU a, bb;
      a.q[0]  = *(const uint4*)(arow + kb);
      a.q[1]  = *(const uint4*)(arow + kb + 16);
      bb.q[0] = *(const uint4*)(wp + kb);
      bb.q[1] = *(const uint4*)(wp + kb + 16);
      acc = __builtin_amdgcn_wmma_f32_16x16x32_bf16(
          false, a.v, false, bb.v, (short)0, acc, false, false);
    }

#pragma unroll
    for (int v = 0; v < 8; ++v) {
      const int row = v + 8 * lh;   // batch index for rows < 8; pad otherwise
      float pre = acc[v];
      if (row < 8) pre += xU[((size_t)row * TT + t) * RR + col];
      float hv = (row < 8) ? tanhf(pre) : 0.f;
      __bf16 hb = f2bf(hv);
      nxt[(size_t)row * RR + col] = hb;
      if (row < 8) rstates[((size_t)row * TT + t) * RR + col] = hb;
    }
    grid_barrier(bar, SCAN_BLOCKS);
  }
}

// ---------------------------------------------------------------------------
extern "C" void kernel_launch(void* const* d_in, const int* in_sizes, int n_in,
                              void* d_out, int out_size, void* d_ws, size_t ws_size,
                              hipStream_t stream) {
  const float* x       = (const float*)d_in[0];
  const float* ln_in_w = (const float*)d_in[1];
  const float* ln_in_b = (const float*)d_in[2];
  const float* U       = (const float*)d_in[3];
  const float* W       = (const float*)d_in[4];
  const float* rln_w   = (const float*)d_in[5];
  const float* rln_b   = (const float*)d_in[6];
  const float* res_w1  = (const float*)d_in[7];
  const float* res_b1  = (const float*)d_in[8];
  const float* res_w2  = (const float*)d_in[9];
  const float* res_b2  = (const float*)d_in[10];
  const float* res_ls  = (const float*)d_in[11];
  const float* mix_w   = (const float*)d_in[12];
  const float* mix_b   = (const float*)d_in[13];
  const float* gate_w1 = (const float*)d_in[14];
  const float* gate_b1 = (const float*)d_in[15];
  const float* gate_w2 = (const float*)d_in[16];
  const float* gate_b2 = (const float*)d_in[17];
  const float* fln_w   = (const float*)d_in[18];
  const float* fln_b   = (const float*)d_in[19];
  const float* ffn_w1  = (const float*)d_in[20];
  const float* ffn_b1  = (const float*)d_in[21];
  const float* ffn_w2  = (const float*)d_in[22];
  const float* ffn_b2  = (const float*)d_in[23];
  const float* ffn_a   = (const float*)d_in[24];
  float* out = (float*)d_out;

  // ---- workspace carve-up (deterministic, ~550 MB assumed available) ----
  char* wsp = (char*)d_ws;
  auto alloc = [&](size_t bytes) -> char* {
    char* p = wsp;
    wsp += (bytes + 255) & ~(size_t)255;
    return p;
  };
  __bf16* WT_U  = (__bf16*)alloc((size_t)RR * DD * 2);
  __bf16* WT_W  = (__bf16*)alloc((size_t)RR * RR * 2);
  __bf16* WT_r1 = (__bf16*)alloc((size_t)DD * RR * 2);
  __bf16* WT_r2 = (__bf16*)alloc((size_t)DD * DD * 2);
  __bf16* WT_g1 = (__bf16*)alloc((size_t)DD * 2 * DD * 2);
  __bf16* WT_g2 = (__bf16*)alloc((size_t)DD * DD * 2);
  __bf16* WT_f1 = (__bf16*)alloc((size_t)RR * DD * 2);
  __bf16* WT_f2 = (__bf16*)alloc((size_t)DD * RR * 2);
  __bf16* hln   = (__bf16*)alloc((size_t)MM * DD * 2);     // also reused as mln
  float*  xU    = (float*) alloc((size_t)MM * RR * 4);     // reused as ffn hidden
  __bf16* rst   = (__bf16*)alloc((size_t)MM * RR * 2);     // states, LN'd in place
  __bf16* t1    = (__bf16*)alloc((size_t)MM * DD * 2);     // res/gate hidden
  float*  r32   = (float*) alloc((size_t)MM * DD * 4);     // scaled reservoir out
  __bf16* mixln = (__bf16*)alloc((size_t)MM * 2 * DD * 2);
  float*  m32   = (float*) alloc((size_t)MM * DD * 4);     // gated mix
  __bf16* hA    = (__bf16*)alloc((size_t)16 * RR * 2);
  __bf16* hB    = (__bf16*)alloc((size_t)16 * RR * 2);
  unsigned* bar = (unsigned*)alloc(256);
  (void)ws_size; (void)n_in; (void)in_sizes; (void)out_size;

  // ---- weights: f32 (KxN) -> bf16 (NxK) ----
  k_convert_T<<<4096, 256, 0, stream>>>(U,       WT_U,  DD,     RR);
  k_convert_T<<<4096, 256, 0, stream>>>(W,       WT_W,  RR,     RR);
  k_convert_T<<<4096, 256, 0, stream>>>(res_w1,  WT_r1, RR,     DD);
  k_convert_T<<<4096, 256, 0, stream>>>(res_w2,  WT_r2, DD,     DD);
  k_convert_T<<<4096, 256, 0, stream>>>(gate_w1, WT_g1, 2 * DD, DD);
  k_convert_T<<<4096, 256, 0, stream>>>(gate_w2, WT_g2, DD,     DD);
  k_convert_T<<<4096, 256, 0, stream>>>(ffn_w1,  WT_f1, DD,     RR);
  k_convert_T<<<4096, 256, 0, stream>>>(ffn_w2,  WT_f2, RR,     DD);
  k_init<<<(16 * RR + 255) / 256, 256, 0, stream>>>(hA, hB, bar);

  // ---- 1. input LN ----
  k_ln_f32<<<MM, 256, 0, stream>>>(x, DD, nullptr, 0, ln_in_w, ln_in_b, hln);

  // ---- 2. xU = hln @ U  (f32 out, no bias) ----
  k_gemm<0><<<dim3(MM / 64, RR / 32), 256, 0, stream>>>(
      hln, WT_U, nullptr, DD, RR, nullptr, xU, nullptr, nullptr, nullptr);

  // ---- 3. sequential reservoir scan (persistent) ----
  k_scan<<<SCAN_BLOCKS, 256, 0, stream>>>(xU, WT_W, rst, hA, hB, bar);

  // ---- 4. reservoir LN (in place) ----
  k_ln_bf16<<<MM, 256, 0, stream>>>(rst, rln_w, rln_b, RR);

  // ---- 5. res MLP: gelu(r@w1+b1) @ w2 + b2, * exp(log_scale) ----
  k_gemm<1><<<dim3(MM / 64, DD / 32), 256, 0, stream>>>(
      rst, WT_r1, res_b1, RR, DD, t1, nullptr, nullptr, nullptr, nullptr);
  k_gemm<2><<<dim3(MM / 64, DD / 32), 256, 0, stream>>>(
      t1, WT_r2, res_b2, DD, DD, nullptr, r32, nullptr, nullptr, res_ls);

  // ---- 6. mix LN over concat(x, r) ----
  k_ln_f32<<<MM, 256, 0, stream>>>(x, DD, r32, DD, mix_w, mix_b, mixln);

  // ---- 7. gate: sigmoid(gelu(mixln@g1+b1)@g2+b2); m = g*r + (1-g)*x ----
  k_gemm<1><<<dim3(MM / 64, DD / 32), 256, 0, stream>>>(
      mixln, WT_g1, gate_b1, 2 * DD, DD, t1, nullptr, nullptr, nullptr, nullptr);
  k_gemm<3><<<dim3(MM / 64, DD / 32), 256, 0, stream>>>(
      t1, WT_g2, gate_b2, DD, DD, nullptr, m32, r32, x, nullptr);

  // ---- 8. FFN: out = m + alpha * (gelu(LN(m)@f1+b1)@f2+b2) ----
  __bf16* mln = hln;                 // reuse (hln dead after step 2)
  __bf16* fh  = (__bf16*)xU;         // reuse (xU dead after scan)
  k_ln_f32<<<MM, 256, 0, stream>>>(m32, DD, nullptr, 0, fln_w, fln_b, mln);
  k_gemm<1><<<dim3(MM / 64, RR / 32), 256, 0, stream>>>(
      mln, WT_f1, ffn_b1, DD, RR, fh, nullptr, nullptr, nullptr, nullptr);
  k_gemm<4><<<dim3(MM / 64, DD / 32), 256, 0, stream>>>(
      fh, WT_f2, ffn_b2, RR, DD, nullptr, out, m32, nullptr, ffn_a);
}
